// ExpansionContrastModule_83494164234797
// MI455X (gfx1250) — matrix operand order repaired and checked by reading, not compiled, not run on hardware
//
#include <hip/hip_runtime.h>
#include <hip/hip_bf16.h>
#include <math.h>

// ---------------- problem constants ----------------
#define NB    4          // batch
#define NCH   64         // channels
#define SHD   2          // heads
#define HC    32         // channels per head
#define NL    8          // directional diffs
#define WID   128
#define HGT   128
#define AREA  16384      // WID*HGT
#define KCH   288        // (NL+1)*HC
#define KEFF  256        // NL*HC (only these key rows survive w[..., :256])
#define EPS_F 1e-5f

// ---------------- WMMA types ----------------
typedef __attribute__((ext_vector_type(16))) __bf16 v16bf;
typedef __attribute__((ext_vector_type(8)))  __bf16 v8bf;
typedef __attribute__((ext_vector_type(8)))  float  v8f;
typedef __attribute__((ext_vector_type(4)))  unsigned int u32x4;

// ---------------- bf16 helpers ----------------
__device__ __forceinline__ unsigned short f2bfu(float f) {
    unsigned int u = __builtin_bit_cast(unsigned int, f);
    u += 0x7FFFu + ((u >> 16) & 1u);       // round-to-nearest-even
    return (unsigned short)(u >> 16);
}
__device__ __forceinline__ __bf16 f2bf(float f) {
    unsigned short h = f2bfu(f);
    return __builtin_bit_cast(__bf16, h);
}
__device__ __forceinline__ __bf16 u2bf(unsigned short h) {
    return __builtin_bit_cast(__bf16, h);
}
// 16 contiguous bf16 from two 8-element (16B-aligned) chunks
__device__ __forceinline__ v16bf load16u(const unsigned short* p0, const unsigned short* p1) {
    u32x4 a = *(const u32x4*)p0;
    u32x4 b = *(const u32x4*)p1;
    v8bf lo = __builtin_bit_cast(v8bf, a);
    v8bf hi = __builtin_bit_cast(v8bf, b);
    return __builtin_shufflevector(lo, hi, 0,1,2,3,4,5,6,7,8,9,10,11,12,13,14,15);
}

// ---------------- WMMA fragment loaders ----------------
// A fragment: 16x32 (MxK), row-major A. Lane l holds row m0+(l&15);
// element e -> K = k0 + ((l&16)?16:0) + e   (contiguous).
__device__ __forceinline__ v16bf load_a_frag(const float* A, int lda, int m0, int k0, int M, int lane) {
    int m = m0 + (lane & 15); if (m > M - 1) m = M - 1;
    const float* row = A + (size_t)m * lda;
    int kb = k0 + ((lane & 16) ? 16 : 0);
    v16bf a;
#pragma unroll
    for (int e = 0; e < 16; ++e) {
        int k = kb + e;
        a[e] = f2bf(row[k]);
    }
    return a;
}
__device__ __forceinline__ v16bf load_a_frag(const unsigned short* A, int lda, int m0, int k0, int M, int lane) {
    int m = m0 + (lane & 15); if (m > M - 1) m = M - 1;
    const unsigned short* row = A + (size_t)m * lda;
    int kb = k0 + ((lane & 16) ? 16 : 0);
    return load16u(row + kb, row + kb + 8);
}
// B fragment (NN): 32x16 (KxN), row-major B[K x N]. Lane l holds col n0+(l&15);
// element e -> K = k0 + ((l&16)?16:0) + e.
__device__ __forceinline__ v16bf load_b_nn(const float* Bm, int ldb, int n0, int k0, int N, int lane) {
    int n = n0 + (lane & 15); if (n > N - 1) n = N - 1;
    int kb = k0 + ((lane & 16) ? 16 : 0);
    v16bf b;
#pragma unroll
    for (int e = 0; e < 16; ++e)
        b[e] = f2bf(Bm[(size_t)(kb + e) * ldb + n]);
    return b;
}
__device__ __forceinline__ v16bf load_b_nn(const unsigned short* Bm, int ldb, int n0, int k0, int N, int lane) {
    int n = n0 + (lane & 15); if (n > N - 1) n = N - 1;
    int kb = k0 + ((lane & 16) ? 16 : 0);
    v16bf b;
#pragma unroll
    for (int e = 0; e < 16; ++e)
        b[e] = u2bf(Bm[(size_t)(kb + e) * ldb + n]);
    return b;
}
// B fragment (NT): B given as Bt[N x K] row-major; B[k][n] = Bt[n][k] (contiguous in k).
__device__ __forceinline__ v16bf load_b_nt(const float* Bt, int ldb, int n0, int k0, int lane) {
    const float* row = Bt + (size_t)(n0 + (lane & 15)) * ldb;
    int kb = k0 + ((lane & 16) ? 16 : 0);
    v16bf b;
#pragma unroll
    for (int e = 0; e < 16; ++e)
        b[e] = f2bf(row[kb + e]);
    return b;
}
__device__ __forceinline__ v16bf load_b_nt(const unsigned short* Bt, int ldb, int n0, int k0, int lane) {
    const unsigned short* row = Bt + (size_t)(n0 + (lane & 15)) * ldb;
    int kb = k0 + ((lane & 16) ? 16 : 0);
    return load16u(row + kb, row + kb + 8);
}

// ---------------- GEMM NN (one 16x16 tile per wave, 4 waves/block) ----------------
// C[M x N] = A[M x K] * B[K x N]; A,B row-major; K % 32 == 0.
template <typename TA, typename TB>
__global__ void __launch_bounds__(128) gemm_nn_kernel(
    const TA* __restrict__ A, size_t sA, int lda,
    const TB* __restrict__ Bm, size_t sB, int ldb,
    float* __restrict__ C, size_t sC, int ldc,
    int M, int N, int K)
{
    int lane = threadIdx.x & 31;
    int wave = threadIdx.x >> 5;
    int n0 = (blockIdx.x * 4 + wave) * 16;
    int m0 = blockIdx.y * 16;
    if (n0 >= N || m0 >= M) return;           // wave-uniform (no barriers here)
    const TA* Ab = A  + (size_t)blockIdx.z * sA;
    const TB* Bb = Bm + (size_t)blockIdx.z * sB;
    float*    Cb = C  + (size_t)blockIdx.z * sC;

    v8f acc = {0.f,0.f,0.f,0.f,0.f,0.f,0.f,0.f};
    for (int k0 = 0; k0 < K; k0 += 32) {
        v16bf a = load_a_frag(Ab, lda, m0, k0, M, lane);
        v16bf b = load_b_nn(Bb, ldb, n0, k0, N, lane);
        acc = __builtin_amdgcn_wmma_f32_16x16x32_bf16(
            false, a, false, b, (short)0, acc, false, false);
    }
    int col = n0 + (lane & 15);
    int rb  = m0 + ((lane & 16) ? 8 : 0);
    if (m0 + 16 <= M && n0 + 16 <= N) {       // fast path: full interior tile
        float* cp = Cb + (size_t)rb * ldc + col;
#pragma unroll
        for (int r = 0; r < 8; ++r) cp[(size_t)r * ldc] = acc[r];
    } else {
#pragma unroll
        for (int r = 0; r < 8; ++r) {
            int row = rb + r;
            if (row < M && col < N) Cb[(size_t)row * ldc + col] = acc[r];
        }
    }
}

// ---------------- GEMM NT (SYRK-style, LDS-staged A, double-buffered) ----------------
// C[M x N] = A[M x K] * Bt[N x K]^T ; contiguous-K contraction.
// Requires M % 16 == 0, N % 16 == 0, K % 32 == 0. All 4 waves of a block share the
// same m0, so the 16x32 A slab is staged once per K-step into LDS (double buffer,
// single barrier per step). Edge waves clamp n0 (wave-uniform) instead of exiting
// so every wave reaches the barrier and EXEC is all-ones around the WMMA.
template <typename TA, typename TB>
__global__ void __launch_bounds__(128) gemm_nt_kernel(
    const TA* __restrict__ A, size_t sA, int lda,
    const TB* __restrict__ Bt, size_t sB, int ldb,
    float* __restrict__ C, size_t sC, int ldc,
    int M, int N, int K)
{
    __shared__ TA As[2][16 * 32];
    int tid  = threadIdx.x;
    int lane = tid & 31;
    int wave = tid >> 5;
    int n0 = (blockIdx.x * 4 + wave) * 16;
    int m0 = blockIdx.y * 16;
    bool store_ok = (n0 + 16 <= N);
    int n0e = store_ok ? n0 : (N - 16);       // wave-uniform clamp (N % 16 == 0)

    const TA* Ab = A  + (size_t)blockIdx.z * sA;
    const TB* Bb = Bt + (size_t)blockIdx.z * sB;
    float*    Cb = C  + (size_t)blockIdx.z * sC;

    // cooperative A staging: 128 threads x 4 elements = 16 rows x 32 K
    int cr = tid >> 3;                        // 0..15 (row within tile)
    int cc = (tid & 7) * 4;                   // 0,4,...,28 (K within tile)
    const TA* Arow = Ab + (size_t)(m0 + cr) * lda + cc;
    int nb_row = n0e + (lane & 15);           // B prefetch row for this lane

    v8f acc = {0.f,0.f,0.f,0.f,0.f,0.f,0.f,0.f};
    for (int k0 = 0; k0 < K; k0 += 32) {
        TA* dst = &As[(k0 >> 5) & 1][cr * 32 + cc];
        const TA* src = Arow + k0;
#pragma unroll
        for (int e = 0; e < 4; ++e) dst[e] = src[e];
        __syncthreads();
        if (k0 + 32 < K) {                    // prefetch next B K-slab
            __builtin_prefetch(Bb + (size_t)nb_row * ldb + (k0 + 32), 0, 1);
        }
        v16bf a = load_a_frag((const TA*)As[(k0 >> 5) & 1], 32, 0, 0, 16, lane);
        v16bf b = load_b_nt(Bb, ldb, n0e, k0, lane);
        acc = __builtin_amdgcn_wmma_f32_16x16x32_bf16(
            false, a, false, b, (short)0, acc, false, false);
    }
    if (store_ok) {
        int col = n0e + (lane & 15);
        int rb  = m0 + ((lane & 16) ? 8 : 0);
        float* cp = Cb + (size_t)rb * ldc + col;
#pragma unroll
        for (int r = 0; r < 8; ++r) cp[(size_t)r * ldc] = acc[r];
    }
}

// ---------------- BatchNorm stats / apply ----------------
__global__ void __launch_bounds__(256) channel_stats_kernel(
    const float* __restrict__ X, float* __restrict__ mean, float* __restrict__ rstd,
    int Cch, int nb, int area)
{
    int c = blockIdx.x, tid = threadIdx.x;
    float s = 0.f, s2 = 0.f;
    for (int b = 0; b < nb; ++b) {
        const float* p = X + ((size_t)b * Cch + c) * area;
        for (int a = tid; a < area; a += 256) { float v = p[a]; s += v; s2 += v * v; }
    }
    __shared__ float r1[256], r2[256];
    r1[tid] = s; r2[tid] = s2; __syncthreads();
    for (int off = 128; off > 0; off >>= 1) {
        if (tid < off) { r1[tid] += r1[tid + off]; r2[tid] += r2[tid + off]; }
        __syncthreads();
    }
    if (tid == 0) {
        float n = (float)nb * (float)area;
        float m = r1[0] / n;
        float v = r2[0] / n - m * m;
        mean[c] = m;
        rstd[c] = rsqrtf(fmaxf(v, 0.f) + EPS_F);
    }
}

__global__ void __launch_bounds__(256) bn_relu_kernel(
    const float* __restrict__ in, float* __restrict__ out,
    const float* __restrict__ mean, const float* __restrict__ rstd,
    const float* __restrict__ g, const float* __restrict__ bb,
    int Cch, int area, size_t total)
{
    size_t i = (size_t)blockIdx.x * 256 + threadIdx.x;
    if (i >= total) return;
    int c = (int)((i / area) % Cch);
    float v = (in[i] - mean[c]) * rstd[c] * g[c] + bb[c];
    out[i] = fmaxf(v, 0.f);
}

// ---------------- build U = [sur(256); cen(32)] in bf16, per (b,s) ----------------
__global__ void __launch_bounds__(256) build_u_kernel(
    const float* __restrict__ y, unsigned short* __restrict__ U)
{
    size_t idx = (size_t)blockIdx.x * 256 + threadIdx.x;   // NB*SHD*HC*AREA threads
    int a = (int)(idx & (AREA - 1));
    int c = (int)((idx >> 14) & 31);
    int s = (int)((idx >> 19) & 1);
    int b = (int)(idx >> 20);
    int i = a >> 7, j = a & 127;
    int d = (s == 0) ? 1 : 2;      // SHIFTS = (1, 2)
    const float* yc = y + ((size_t)b * NCH + s * HC + c) * AREA;
    float v = yc[a];
    unsigned short* Ub = U + (size_t)(b * SHD + s) * KCH * AREA;
    const int DI[8] = {-1,-1,-1, 0, 1, 1, 1, 0};
    const int DJ[8] = {-1, 0, 1, 1, 1, 0,-1,-1};
#pragma unroll
    for (int l = 0; l < 8; ++l) {
        int ii = i + DI[l] * d, jj = j + DJ[l] * d;
        float nbv = (ii >= 0 && ii < HGT && jj >= 0 && jj < WID) ? yc[ii * WID + jj] : 0.f;
        Ub[(size_t)(l * HC + c) * AREA + a] = f2bfu(v - nbv);
    }
    Ub[(size_t)(KEFF + c) * AREA + a] = f2bfu(v);
}

// ---------------- per-(b,s) channel norms from Gram-derived matrices ----------------
// kn[k]^2 = sum_j KW[k,j]*N1[k,j]  (N1 = KW*G),  qn[c]^2 = sum_j' QW[c,j']*M1[c,256+j']
__global__ void norms_kernel(
    const float* __restrict__ M1, const float* __restrict__ N1,
    const float* __restrict__ qw, const float* __restrict__ kw,
    float* __restrict__ qn, float* __restrict__ kn)
{
    int bs = blockIdx.x, s = bs & 1, tid = threadIdx.x;
    if (tid < KEFF) {
        const float* kwr = kw + (size_t)s * KCH * KCH + (size_t)tid * KCH;
        const float* n1r = N1 + (size_t)bs * KEFF * KCH + (size_t)tid * KCH;
        float acc = 0.f;
        for (int j = 0; j < KCH; ++j) acc += kwr[j] * n1r[j];
        kn[bs * KEFF + tid] = fmaxf(sqrtf(fmaxf(acc, 0.f)), 1e-12f);
    } else if (tid < KCH) {
        int c = tid - KEFF;
        const float* qwr = qw + (size_t)s * HC * HC + (size_t)c * HC;
        const float* m1r = M1 + (size_t)bs * HC * KCH + (size_t)c * KCH;
        float acc = 0.f;
        for (int j = 0; j < HC; ++j) acc += qwr[j] * m1r[KEFF + j];
        qn[bs * HC + c] = fmaxf(sqrtf(fmaxf(acc, 0.f)), 1e-12f);
    }
}

// ---------------- instance-norm + softmax over [32 x 256] per (b,s) ----------------
__global__ void __launch_bounds__(256) inorm_softmax_kernel(
    const float* __restrict__ wun,
    const float* __restrict__ qn, const float* __restrict__ kn,
    float* __restrict__ swout)
{
    __shared__ float sh[HC * KEFF];         // 32 KB
    __shared__ float r1[256], r2[256], rr[256];
    __shared__ float rowm[HC], rows[HC];
    int bs = blockIdx.x, tid = threadIdx.x;
    const float inv_sqrt_area = 1.0f / 128.0f;   // 1/sqrt(16384)
    float s1 = 0.f, s2 = 0.f;
    for (int idx = tid; idx < HC * KEFF; idx += 256) {
        int c = idx >> 8, k = idx & 255;
        float v = wun[(size_t)bs * HC * KEFF + idx] * inv_sqrt_area
                  / (qn[bs * HC + c] * kn[bs * KEFF + k]);
        sh[idx] = v; s1 += v; s2 += v * v;
    }
    r1[tid] = s1; r2[tid] = s2; __syncthreads();
    for (int off = 128; off > 0; off >>= 1) {
        if (tid < off) { r1[tid] += r1[tid + off]; r2[tid] += r2[tid + off]; }
        __syncthreads();
    }
    float m = r1[0] / (float)(HC * KEFF);
    float var = r2[0] / (float)(HC * KEFF) - m * m;
    float rstd = rsqrtf(fmaxf(var, 0.f) + EPS_F);

    int row = tid >> 3, sub = tid & 7;      // 8 threads per row
    float mx = -3.4e38f;
    for (int k = sub; k < KEFF; k += 8)
        mx = fmaxf(mx, (sh[row * KEFF + k] - m) * rstd);
    rr[tid] = mx; __syncthreads();
    if (sub == 0) { float v = rr[tid]; for (int t = 1; t < 8; ++t) v = fmaxf(v, rr[tid + t]); rowm[row] = v; }
    __syncthreads();
    float rmax = rowm[row], se = 0.f;
    for (int k = sub; k < KEFF; k += 8)
        se += __expf((sh[row * KEFF + k] - m) * rstd - rmax);
    rr[tid] = se; __syncthreads();
    if (sub == 0) { float v = 0.f; for (int t = 0; t < 8; ++t) v += rr[tid + t]; rows[row] = v; }
    __syncthreads();
    float inv = 1.0f / rows[row];
    for (int k = sub; k < KEFF; k += 8) {
        float v = (sh[row * KEFF + k] - m) * rstd;
        swout[(size_t)bs * HC * KEFF + row * KEFF + k] = __expf(v - rmax) * inv;
    }
}

// ---------------- launcher ----------------
extern "C" void kernel_launch(void* const* d_in, const int* in_sizes, int n_in,
                              void* d_out, int out_size, void* d_ws, size_t ws_size,
                              hipStream_t stream)
{
    (void)in_sizes; (void)n_in; (void)out_size; (void)ws_size;
    const float* x       = (const float*)d_in[0];
    const float* trans_w = (const float*)d_in[1];
    const float* bn1_g   = (const float*)d_in[2];
    const float* bn1_b   = (const float*)d_in[3];
    const float* query_w = (const float*)d_in[4];
    const float* key_w   = (const float*)d_in[5];
    const float* out_w   = (const float*)d_in[6];
    const float* bn2_g   = (const float*)d_in[7];
    const float* bn2_b   = (const float*)d_in[8];
    float* outp = (float*)d_out;

    // workspace carve-up (256B aligned)
    char* cur = (char*)d_ws;
    auto carve = [&](size_t bytes) -> void* {
        void* p = (void*)cur;
        cur += (bytes + 255) & ~(size_t)255;
        return p;
    };
    const size_t actN = (size_t)NB * NCH * AREA;                 // 4.19M f32
    float*          ws_y   = (float*)carve(actN * 4);
    float*          ws_att = (float*)carve(actN * 4);
    float*          ws_z   = (float*)carve(actN * 4);
    unsigned short* ws_U   = (unsigned short*)carve((size_t)NB * SHD * KCH * AREA * 2);
    float*          ws_G   = (float*)carve((size_t)NB * SHD * KCH * KCH * 4);
    float*          ws_N1  = (float*)carve((size_t)NB * SHD * KEFF * KCH * 4);
    float*          ws_M1  = (float*)carve((size_t)NB * SHD * HC * KCH * 4);
    float*          ws_wun = (float*)carve((size_t)NB * SHD * HC * KEFF * 4);
    float*          ws_sw  = (float*)carve((size_t)NB * SHD * HC * KEFF * 4);
    float*          ws_qn  = (float*)carve((size_t)NB * SHD * HC * 4);
    float*          ws_kn  = (float*)carve((size_t)NB * SHD * KEFF * 4);
    float*          mean1  = (float*)carve(NCH * 4);
    float*          rstd1  = (float*)carve(NCH * 4);
    float*          mean2  = (float*)carve(NCH * 4);
    float*          rstd2  = (float*)carve(NCH * 4);

    // 1) y_pre[b] = trans_w(64x64) * x[b](64xAREA)
    gemm_nn_kernel<float, float><<<dim3(AREA / 64, NCH / 16, NB), 128, 0, stream>>>(
        trans_w, 0, NCH, x, (size_t)NCH * AREA, AREA,
        ws_y, (size_t)NCH * AREA, AREA, NCH, AREA, NCH);
    // 2) BN1 stats + apply (in place)
    channel_stats_kernel<<<NCH, 256, 0, stream>>>(ws_y, mean1, rstd1, NCH, NB, AREA);
    bn_relu_kernel<<<(unsigned)(actN / 256), 256, 0, stream>>>(
        ws_y, ws_y, mean1, rstd1, bn1_g, bn1_b, NCH, AREA, actN);
    // 3) build U (sur diffs + cen) in bf16
    build_u_kernel<<<(unsigned)((size_t)NB * SHD * HC * AREA / 256), 256, 0, stream>>>(ws_y, ws_U);
    // 4) Gram: G[bs] = U * U^T  (288x288, K=16384) -- the heavy WMMA SYRK
    gemm_nt_kernel<unsigned short, unsigned short>
        <<<dim3(5, KCH / 16, NB * SHD), 128, 0, stream>>>(
        ws_U, (size_t)KCH * AREA, AREA, ws_U, (size_t)KCH * AREA, AREA,
        ws_G, (size_t)KCH * KCH, KCH, KCH, KCH, AREA);
    // 5) M1[bs] = QW[s](32x32) * G[bs][256:,:](32x288);  N1[bs] = KW[s](256x288) * G[bs](288x288)
    for (int s = 0; s < SHD; ++s) {
        gemm_nn_kernel<float, float><<<dim3(5, HC / 16, NB), 128, 0, stream>>>(
            query_w + (size_t)s * HC * HC, 0, HC,
            ws_G + (size_t)s * KCH * KCH + (size_t)KEFF * KCH, (size_t)SHD * KCH * KCH, KCH,
            ws_M1 + (size_t)s * HC * KCH, (size_t)SHD * HC * KCH, KCH,
            HC, KCH, HC);
        gemm_nn_kernel<float, float><<<dim3(5, KEFF / 16, NB), 128, 0, stream>>>(
            key_w + (size_t)s * KCH * KCH, 0, KCH,
            ws_G + (size_t)s * KCH * KCH, (size_t)SHD * KCH * KCH, KCH,
            ws_N1 + (size_t)s * KEFF * KCH, (size_t)SHD * KEFF * KCH, KCH,
            KEFF, KCH, KCH);
    }
    // 6) per-channel q/k L2 norms
    norms_kernel<<<NB * SHD, KCH, 0, stream>>>(ws_M1, ws_N1, query_w, key_w, ws_qn, ws_kn);
    // 7) w_un[bs] = M1(32x288) * KW[s](256x288)^T
    for (int s = 0; s < SHD; ++s) {
        gemm_nt_kernel<float, float><<<dim3(KEFF / 64, HC / 16, NB), 128, 0, stream>>>(
            ws_M1 + (size_t)s * HC * KCH, (size_t)SHD * HC * KCH, KCH,
            key_w + (size_t)s * KCH * KCH, 0, KCH,
            ws_wun + (size_t)s * HC * KEFF, (size_t)SHD * HC * KEFF, KEFF,
            HC, KEFF, KCH);
    }
    // 8) scale by 1/(|q||k|sqrt(AREA)), instance-norm, softmax
    inorm_softmax_kernel<<<NB * SHD, 256, 0, stream>>>(ws_wun, ws_qn, ws_kn, ws_sw);
    // 9) att_out[bs] = sw(32x256) * sur(256xAREA)  -> [B,64,AREA] contiguous
    gemm_nn_kernel<float, unsigned short><<<dim3(AREA / 64, HC / 16, NB * SHD), 128, 0, stream>>>(
        ws_sw, (size_t)HC * KEFF, KEFF,
        ws_U, (size_t)KCH * AREA, AREA,
        ws_att, (size_t)HC * AREA, AREA, HC, AREA, KEFF);
    // 10) z_pre[b] = out_w(64x64) * att[b](64xAREA)
    gemm_nn_kernel<float, float><<<dim3(AREA / 64, NCH / 16, NB), 128, 0, stream>>>(
        out_w, 0, NCH, ws_att, (size_t)NCH * AREA, AREA,
        ws_z, (size_t)NCH * AREA, AREA, NCH, AREA, NCH);
    // 11) BN2 stats + apply -> d_out
    channel_stats_kernel<<<NCH, 256, 0, stream>>>(ws_z, mean2, rstd2, NCH, NB, AREA);
    bn_relu_kernel<<<(unsigned)(actN / 256), 256, 0, stream>>>(
        ws_z, outp, mean2, rstd2, bn2_g, bn2_b, NCH, AREA, actN);
}
